// ContextLSTMCore_67224828117480
// MI455X (gfx1250) — matrix operand order repaired and checked by reading, not compiled, not run on hardware
//
#include <hip/hip_runtime.h>
#include <hip/hip_bf16.h>

// ---------------------------------------------------------------------------
// ContextLSTM core for MI455X (gfx1250, wave32, WMMA).
// All matmuls run as bf16 WMMA (v_wmma_f32_16x16x32_bf16) with f32 accum.
// Weights are converted once per call into B-matrix wave32 lane layout
// (ISA 7.12.2) and stay resident in the 192MB L2 (77MB bf16 total).
// Activations are packed per step (k-major) and staged through LDS with a
// double buffer: the 4KB A chunk per k-tile is loaded once per block
// (128 lanes x 2 b128, coalesced) instead of once per wave.
// ---------------------------------------------------------------------------

typedef __attribute__((ext_vector_type(16))) __bf16 v16bf;
typedef __attribute__((ext_vector_type(8)))  float  v8f;

#define ACT_NONE 0
#define ACT_ELU  1
#define ACT_TANH 2
#define ACT_SIG  3

static constexpr int T_   = 64;
static constexpr int B_   = 64;
static constexpr int ACTD = 6;
static constexpr int EMB  = 1536;
static constexpr int DET  = 2048;
static constexpr int CTX  = 512;
static constexpr int HID  = 1024;
static constexpr int ZF   = 1024;

__device__ __forceinline__ float sigf(float x) { return 1.0f / (1.0f + __expf(-x)); }

__device__ __forceinline__ float apply_act(float x, int act) {
    switch (act) {
        case ACT_ELU:  return x > 0.0f ? x : (__expf(x) - 1.0f);
        case ACT_TANH: return tanhf(x);
        case ACT_SIG:  return sigf(x);
        default:       return x;
    }
}

// ---------------------------------------------------------------------------
// Pack a (64 x Ksum) fp32 activation matrix (concat of up to 3 sources) into
// bf16 A-tiles in the 16-bit A-matrix 16x32 wave32 lane layout:
//   lane 0-15 : M=lane,    VGPR v holds K = (v<4?0:16) + (v%4)*2 + p
//   lane16-31 : M=lane-16, same + 8
// K-MAJOR tile order: tile = (ktile * 4 + mtile), 512 bf16 per tile,
// elem = lane*16 + v*2 + p.  One k-tile = 4 contiguous tiles = 4KB.
// ---------------------------------------------------------------------------
__global__ void __launch_bounds__(256) pack_a_kernel(
    const float* __restrict__ S0, int K0,
    const float* __restrict__ S1, int K1,
    const float* __restrict__ S2, int K2,
    int KT, __bf16* __restrict__ dst)
{
    int e = blockIdx.x * 256 + threadIdx.x;
    int total = 4 * KT * 512;
    if (e >= total) return;
    int tile = e >> 9, r = e & 511;
    int kt = tile >> 2, mt = tile & 3;                 // k-major
    int lane = r >> 4, q = r & 15, v = q >> 1, p = q & 1;
    int m  = mt * 16 + (lane & 15);
    int kk = ((v < 4) ? 0 : 16) + ((lane >= 16) ? 8 : 0) + (v & 3) * 2 + p;
    int k  = kt * 32 + kk;
    float val = 0.0f;
    if (k < K0)                 val = S0[(long)m * K0 + k];
    else if (k < K0 + K1)       val = S1[(long)m * K1 + (k - K0)];
    else if (k < K0 + K1 + K2)  val = S2[(long)m * K2 + (k - K0 - K1)];
    dst[e] = (__bf16)val;
}

// ---------------------------------------------------------------------------
// Pack a (Ksrc x N) fp32 weight matrix segment into bf16 B-tiles in the
// 16-bit B-matrix 32x16 wave32 lane layout (by ISA sparse-B analogy):
//   lane 0-15 : N=lane,    VGPR v holds K = (v<4?0:16) + (v%4)*2 + p
//   lane16-31 : N=lane-16, same + 8
// Destination tile order: (ntile * KTtot + kt_off + ktl).
// Rows k >= Ksrc are zero-padded (used for the 1542->1568 K pad of W_in).
// ---------------------------------------------------------------------------
__global__ void __launch_bounds__(256) pack_b_kernel(
    const float* __restrict__ W, int Ksrc, int N,
    int KTtot, int kt_off, int KTseg, __bf16* __restrict__ dst)
{
    long e = (long)blockIdx.x * 256 + threadIdx.x;
    long total = (long)(N / 16) * KTseg * 512;
    if (e >= total) return;
    long tile = e >> 9; int r = (int)(e & 511);
    int ktl = (int)(tile % KTseg);
    int nt  = (int)(tile / KTseg);
    int lane = r >> 4, q = r & 15, v = q >> 1, p = q & 1;
    int n  = nt * 16 + (lane & 15);
    int kk = ((v < 4) ? 0 : 16) + ((lane >= 16) ? 8 : 0) + (v & 3) * 2 + p;
    int k  = ktl * 32 + kk;
    float val = (k < Ksrc) ? W[(long)k * N + n] : 0.0f;
    dst[((long)(nt * KTtot + kt_off + ktl) << 9) + r] = (__bf16)val;
}

// ---------------------------------------------------------------------------
// WMMA GEMM: out[64, N] = act( A[64, KT*32] @ B[KT*32, N] + bias[N] )
// One wave owns a full 64-row x 16-col strip (4 M-tiles share each B tile:
// 4x reuse of the dominant, L2-resident K-stream). The A chunk for each
// k-tile (4KB) is staged into a double-buffered LDS tile by the whole block
// (128 of its lanes, two coalesced b128 each), so A is fetched once per block per k-tile
// instead of once per wave. Grid = (N/16)/8 blocks of 8 waves.
// ---------------------------------------------------------------------------
__global__ void __launch_bounds__(256) wmma_gemm_kernel(
    const __bf16* __restrict__ Apack, int KT,
    const __bf16* __restrict__ Bpack,
    const float* __restrict__ bias,
    float* __restrict__ out, int N, int act)
{
    __shared__ alignas(32) __bf16 smem[2][4 * 512];   // 8KB double-buffered A stage

    const int tid  = threadIdx.x;
    const int lane = tid & 31;
    const int wave = tid >> 5;
    const int nt   = blockIdx.x * 8 + wave;

    const v16bf* Av = (const v16bf*)Apack;            // 128 x v16bf per k-tile
    const v16bf* bp = (const v16bf*)Bpack + (size_t)nt * KT * 32 + lane;

    // preload k-tile 0 into buffer 0
    if (tid < 128) ((v16bf*)smem[0])[tid] = Av[tid];

    v8f acc0 = {}, acc1 = {}, acc2 = {}, acc3 = {};

    for (int kt = 0; kt < KT; ++kt) {
        __syncthreads();   // s_barrier_signal/s_barrier_wait: buffer kt&1 ready
        if (tid < 128 && kt + 1 < KT)
            ((v16bf*)smem[(kt + 1) & 1])[tid] = Av[(size_t)(kt + 1) * 128 + tid];

        v16bf b = *bp;                                      // 2 x global_load_b128
        __builtin_prefetch((const void*)(bp + 32), 0, 0);   // global_prefetch_b8
        bp += 32;

        const v16bf* sa = (const v16bf*)smem[kt & 1];       // ds_load_b128 x8
        v16bf a0 = sa[0 * 32 + lane];
        v16bf a1 = sa[1 * 32 + lane];
        v16bf a2 = sa[2 * 32 + lane];
        v16bf a3 = sa[3 * 32 + lane];

        acc0 = __builtin_amdgcn_wmma_f32_16x16x32_bf16(false, a0, false, b, (short)0, acc0, false, false);
        acc1 = __builtin_amdgcn_wmma_f32_16x16x32_bf16(false, a1, false, b, (short)0, acc1, false, false);
        acc2 = __builtin_amdgcn_wmma_f32_16x16x32_bf16(false, a2, false, b, (short)0, acc2, false, false);
        acc3 = __builtin_amdgcn_wmma_f32_16x16x32_bf16(false, a3, false, b, (short)0, acc3, false, false);
    }

    // C/D layout (16x16 f32): VGPR r, lanes 0-15 -> M=r, lanes 16-31 -> M=r+8, N=lane%16
    const int n       = nt * 16 + (lane & 15);
    const int mrow_hi = (lane >> 4) * 8;
    const float bv    = bias[n];

#define STORE_TILE(ACC, MT)                                                    \
    {                                                                          \
        _Pragma("unroll")                                                      \
        for (int r = 0; r < 8; ++r) {                                          \
            float vv = apply_act((ACC)[r] + bv, act);                          \
            out[(long)((MT) * 16 + mrow_hi + r) * N + n] = vv;                 \
        }                                                                      \
    }
    STORE_TILE(acc0, 0)
    STORE_TILE(acc1, 1)
    STORE_TILE(acc2, 2)
    STORE_TILE(acc3, 3)
#undef STORE_TILE
}

// ---------------------------------------------------------------------------
// Reset-mask the carried state in place (h, cell, z, ctx).
// ---------------------------------------------------------------------------
__global__ void __launch_bounds__(256) mask_states_kernel(
    const unsigned char* __restrict__ rst,   // [B] bools for this timestep
    float* __restrict__ h, float* __restrict__ cell,
    float* __restrict__ z, float* __restrict__ ctx)
{
    int i = blockIdx.x * 256 + threadIdx.x;
    const int NH = B_ * DET, NZ = B_ * ZF, NC = B_ * CTX;
    if (i < NH)                      { if (rst[i >> 11]) h[i] = 0.0f; }
    else if (i < 2 * NH)             { int j = i - NH;          if (rst[j >> 11]) cell[j] = 0.0f; }
    else if (i < 2 * NH + NZ)        { int j = i - 2 * NH;      if (rst[j >> 10]) z[j]    = 0.0f; }
    else if (i < 2 * NH + NZ + NC)   { int j = i - 2 * NH - NZ; if (rst[j >> 9])  ctx[j]  = 0.0f; }
}

// ---------------------------------------------------------------------------
// LSTM elementwise: g4 = [i | f | c | o] each [B, DET].
// ---------------------------------------------------------------------------
__global__ void __launch_bounds__(256) lstm_elem_kernel(
    const float* __restrict__ g4,
    float* __restrict__ h, float* __restrict__ cell,
    float* __restrict__ deter_out)
{
    int i = blockIdx.x * 256 + threadIdx.x;
    if (i >= B_ * DET) return;
    int b = i >> 11, d = i & (DET - 1);
    const float* g = g4 + (long)b * (4 * DET);
    float ig = g[d], fg = g[DET + d], cg = g[2 * DET + d], og = g[3 * DET + d];
    float c  = sigf(fg) * cell[i] + sigf(ig) * tanhf(cg);
    float hh = sigf(og) * tanhf(c);
    cell[i] = c; h[i] = hh; deter_out[i] = hh;
}

// ---------------------------------------------------------------------------
// Gated context update: ctx = gate*cand + (1-gate)*ctx   (gate pre-sigmoided,
// cand pre-tanh'd by the GEMM epilogues).
// ---------------------------------------------------------------------------
__global__ void __launch_bounds__(256) ctx_update_kernel(
    const float* __restrict__ gate, const float* __restrict__ cand,
    float* __restrict__ ctx, float* __restrict__ ctxs_out)
{
    int i = blockIdx.x * 256 + threadIdx.x;
    if (i >= B_ * CTX) return;
    float g = gate[i];
    float c = g * cand[i] + (1.0f - g) * ctx[i];
    ctx[i] = c; ctxs_out[i] = c;
}

// ---------------------------------------------------------------------------
// Straight-through Gumbel sample. Forward value of
//   hard + probs - stop_grad(probs)  ==  hard (one-hot of argmax(lg + g)).
// Deterministic counter-based hash RNG per (t,b,d,c).
// ---------------------------------------------------------------------------
__device__ __forceinline__ unsigned hash_u32(unsigned x) {
    x ^= x >> 16; x *= 0x7feb352dU;
    x ^= x >> 15; x *= 0x846ca68bU;
    x ^= x >> 16; return x;
}

__global__ void __launch_bounds__(256) gumbel_sample_kernel(
    const float* __restrict__ post_t,      // [B, ZF] logits for this timestep
    float* __restrict__ z, float* __restrict__ stoch_out, int t)
{
    int i = blockIdx.x * 256 + threadIdx.x;   // one thread per (b, stoch_dim)
    if (i >= B_ * 32) return;
    int b = i >> 5, d = i & 31;
    const float* lg = post_t + (long)b * ZF + d * 32;
    float best = -3.4e38f; int bi = 0;
    #pragma unroll 4
    for (int c = 0; c < 32; ++c) {
        unsigned hsh = hash_u32((((unsigned)t * 2048u + (unsigned)i) << 5) + (unsigned)c + 0x9e3779b9u);
        float u  = (float)(hsh >> 8) * (1.0f / 16777216.0f);
        u        = fminf(fmaxf(u, 1e-7f), 1.0f - 1e-7f);
        float gn = -__logf(-__logf(u));
        float s  = lg[c] + gn;                 // TEMP == 1.0
        if (s > best) { best = s; bi = c; }
    }
    float* zr = z         + (long)b * ZF + d * 32;
    float* sr = stoch_out + (long)b * ZF + d * 32;
    for (int c = 0; c < 32; ++c) {
        float vv = (c == bi) ? 1.0f : 0.0f;
        zr[c] = vv; sr[c] = vv;
    }
}

// ---------------------------------------------------------------------------
// features = concat(deter, stoch, ctxs) along last axis, over the whole [T,B].
// ---------------------------------------------------------------------------
__global__ void __launch_bounds__(256) features_kernel(
    const float* __restrict__ deter, const float* __restrict__ stoch,
    const float* __restrict__ ctxs, float* __restrict__ feat)
{
    long i = (long)blockIdx.x * 256 + threadIdx.x;
    const long total = (long)T_ * B_ * (DET + ZF + CTX);
    if (i >= total) return;
    long tb = i / (DET + ZF + CTX);
    int  d  = (int)(i % (DET + ZF + CTX));
    float v;
    if (d < DET)            v = deter[tb * DET + d];
    else if (d < DET + ZF)  v = stoch[tb * ZF + (d - DET)];
    else                    v = ctxs[tb * CTX + (d - DET - ZF)];
    feat[i] = v;
}

// ---------------------------------------------------------------------------
// Host-side orchestration.
// ---------------------------------------------------------------------------
static inline int cdiv(long a, long b) { return (int)((a + b - 1) / b); }

extern "C" void kernel_launch(void* const* d_in, const int* in_sizes, int n_in,
                              void* d_out, int out_size, void* d_ws, size_t ws_size,
                              hipStream_t stream) {
    (void)in_sizes; (void)n_in; (void)out_size; (void)ws_size;

    // ---- inputs (setup_inputs dict order) ----
    const float* embeds  = (const float*)d_in[0];
    const float* actions = (const float*)d_in[1];
    const unsigned char* resets = (const unsigned char*)d_in[2];
    const float* h0      = (const float*)d_in[3];
    const float* z0      = (const float*)d_in[4];
    const float* c0      = (const float*)d_in[5];
    const float* cell0   = (const float*)d_in[6];
    const float* W_in    = (const float*)d_in[7];
    const float* b_in    = (const float*)d_in[8];
    const float* W_ih    = (const float*)d_in[9];
    const float* W_hh    = (const float*)d_in[10];
    const float* b_lstm  = (const float*)d_in[11];
    const float* W_cand  = (const float*)d_in[12];
    const float* b_cand  = (const float*)d_in[13];
    const float* W_gate  = (const float*)d_in[14];
    const float* b_gate  = (const float*)d_in[15];
    const float* W_pr1   = (const float*)d_in[16];
    const float* b_pr1   = (const float*)d_in[17];
    const float* W_pr2   = (const float*)d_in[18];
    const float* b_pr2   = (const float*)d_in[19];
    const float* W_cp1   = (const float*)d_in[20];
    const float* b_cp1   = (const float*)d_in[21];
    const float* W_cp2   = (const float*)d_in[22];
    const float* b_cp2   = (const float*)d_in[23];
    const float* W_po1   = (const float*)d_in[24];
    const float* b_po1   = (const float*)d_in[25];
    const float* W_po2   = (const float*)d_in[26];
    const float* b_po2   = (const float*)d_in[27];

    // ---- output slabs (reference return order, concatenated flat) ----
    const long TB = (long)T_ * B_;
    float* out      = (float*)d_out;
    float* o_priors = out;
    float* o_posts  = o_priors + TB * ZF;
    float* o_cprior = o_posts  + TB * ZF;
    float* o_deter  = o_cprior + TB * ZF;
    float* o_stoch  = o_deter  + TB * DET;
    float* o_ctxs   = o_stoch  + TB * ZF;
    float* o_gates  = o_ctxs   + TB * CTX;
    float* o_feat   = o_gates  + TB * CTX;
    float* o_hf     = o_feat   + TB * (DET + ZF + CTX);
    float* o_zf     = o_hf     + (long)B_ * DET;
    float* o_cf     = o_zf     + (long)B_ * ZF;
    float* o_cellf  = o_cf     + (long)B_ * CTX;

    // ---- workspace bump allocator ----
    size_t off = 0;
    auto walloc = [&](size_t bytes) -> void* {
        void* p = (char*)d_ws + off;
        off += (bytes + 255) & ~(size_t)255;
        return p;
    };

    // bf16 weight packs (K-tiles of 32, N-tiles of 16); 77 MB total -> L2-resident
    const int KT_IN = 49, KT_G = 96, KT_H = 64, KT_HC = 80, KT_HE = 112,
              KT_C = 16, KT_T = 32;
    __bf16* Win_t   = (__bf16*)walloc((size_t)(HID / 16) * KT_IN * 512 * 2);
    __bf16* Wg_t    = (__bf16*)walloc((size_t)(4 * DET / 16) * KT_G * 512 * 2);
    __bf16* Wcand_t = (__bf16*)walloc((size_t)(CTX / 16) * KT_H * 512 * 2);
    __bf16* Wgate_t = (__bf16*)walloc((size_t)(CTX / 16) * KT_HC * 512 * 2);
    __bf16* Wpr1_t  = (__bf16*)walloc((size_t)(HID / 16) * KT_H * 512 * 2);
    __bf16* Wpr2_t  = (__bf16*)walloc((size_t)(ZF / 16) * KT_T * 512 * 2);
    __bf16* Wcp1_t  = (__bf16*)walloc((size_t)(HID / 16) * KT_C * 512 * 2);
    __bf16* Wcp2_t  = (__bf16*)walloc((size_t)(ZF / 16) * KT_T * 512 * 2);
    __bf16* Wpo1_t  = (__bf16*)walloc((size_t)(HID / 16) * KT_HE * 512 * 2);
    __bf16* Wpo2_t  = (__bf16*)walloc((size_t)(ZF / 16) * KT_T * 512 * 2);

    // fp32 state + activation buffers
    float* h_s    = (float*)walloc((size_t)B_ * DET * 4);
    float* cell_s = (float*)walloc((size_t)B_ * DET * 4);
    float* z_s    = (float*)walloc((size_t)B_ * ZF * 4);
    float* ctx_s  = (float*)walloc((size_t)B_ * CTX * 4);
    float* x_b    = (float*)walloc((size_t)B_ * HID * 4);
    float* g4_b   = (float*)walloc((size_t)B_ * 4 * DET * 4);
    float* cand_b = (float*)walloc((size_t)B_ * CTX * 4);
    float* tmp1_b = (float*)walloc((size_t)B_ * HID * 4);

    // bf16 activation packs (k-major tile order)
    __bf16* A1  = (__bf16*)walloc((size_t)4 * KT_IN * 512 * 2);
    __bf16* A2  = (__bf16*)walloc((size_t)4 * KT_G * 512 * 2);
    __bf16* Ah  = (__bf16*)walloc((size_t)4 * KT_H * 512 * 2);
    __bf16* Ahc = (__bf16*)walloc((size_t)4 * KT_HC * 512 * 2);
    __bf16* Ahe = (__bf16*)walloc((size_t)4 * KT_HE * 512 * 2);
    __bf16* Ac  = (__bf16*)walloc((size_t)4 * KT_C * 512 * 2);
    __bf16* At  = (__bf16*)walloc((size_t)4 * KT_T * 512 * 2);

    // ---- one-time (per call) weight conversion into WMMA B layout ----
    auto packB = [&](const float* W, int K, int N, int KTtot, int ktOff, int KTseg, __bf16* dst) {
        long tot = (long)(N / 16) * KTseg * 512;
        pack_b_kernel<<<cdiv(tot, 256), 256, 0, stream>>>(W, K, N, KTtot, ktOff, KTseg, dst);
    };
    packB(W_in,  1542, HID,     KT_IN, 0, KT_IN, Win_t);
    packB(W_ih,  1024, 4 * DET, KT_G,  0, 32,    Wg_t);   // rows 0..1023 of fused [W_ih; W_hh]
    packB(W_hh,  2048, 4 * DET, KT_G, 32, 64,    Wg_t);   // rows 1024..3071
    packB(W_cand, DET, CTX,     KT_H,  0, KT_H,  Wcand_t);
    packB(W_gate, DET + CTX, CTX, KT_HC, 0, KT_HC, Wgate_t);
    packB(W_pr1,  DET, HID,     KT_H,  0, KT_H,  Wpr1_t);
    packB(W_pr2,  HID, ZF,      KT_T,  0, KT_T,  Wpr2_t);
    packB(W_cp1,  CTX, HID,     KT_C,  0, KT_C,  Wcp1_t);
    packB(W_cp2,  HID, ZF,      KT_T,  0, KT_T,  Wcp2_t);
    packB(W_po1,  DET + EMB, HID, KT_HE, 0, KT_HE, Wpo1_t);
    packB(W_po2,  HID, ZF,      KT_T,  0, KT_T,  Wpo2_t);

    // ---- initial carry ----
    hipMemcpyAsync(h_s,    h0,    (size_t)B_ * DET * 4, hipMemcpyDeviceToDevice, stream);
    hipMemcpyAsync(z_s,    z0,    (size_t)B_ * ZF  * 4, hipMemcpyDeviceToDevice, stream);
    hipMemcpyAsync(ctx_s,  c0,    (size_t)B_ * CTX * 4, hipMemcpyDeviceToDevice, stream);
    hipMemcpyAsync(cell_s, cell0, (size_t)B_ * DET * 4, hipMemcpyDeviceToDevice, stream);

    auto packA = [&](const float* s0, int k0, const float* s1, int k1,
                     const float* s2, int k2, int kt, __bf16* dst) {
        pack_a_kernel<<<cdiv((long)4 * kt * 512, 256), 256, 0, stream>>>(s0, k0, s1, k1, s2, k2, kt, dst);
    };
    auto gemm = [&](const __bf16* A, int kt, const __bf16* Bt, const float* bias,
                    float* o, int N, int act) {
        wmma_gemm_kernel<<<(N / 16) / 8, 256, 0, stream>>>(A, kt, Bt, bias, o, N, act);
    };

    // ---- recurrence: serial over T, wide per step ----
    for (int t = 0; t < T_; ++t) {
        const float* emb_t = embeds  + (long)t * B_ * EMB;
        const float* act_t = actions + (long)t * B_ * ACTD;
        const unsigned char* rst_t = resets + (long)t * B_;

        mask_states_kernel<<<cdiv((long)B_ * (2 * DET + ZF + CTX), 256), 256, 0, stream>>>(
            rst_t, h_s, cell_s, z_s, ctx_s);

        // x = elu([z, act, ctx] @ W_in + b_in)
        packA(z_s, ZF, act_t, ACTD, ctx_s, CTX, KT_IN, A1);
        gemm(A1, KT_IN, Win_t, b_in, x_b, HID, ACT_ELU);

        // g4 = [x, h] @ [W_ih; W_hh] + b_lstm ; LSTM update
        packA(x_b, HID, h_s, DET, nullptr, 0, KT_G, A2);
        gemm(A2, KT_G, Wg_t, b_lstm, g4_b, 4 * DET, ACT_NONE);
        lstm_elem_kernel<<<cdiv((long)B_ * DET, 256), 256, 0, stream>>>(
            g4_b, h_s, cell_s, o_deter + (long)t * B_ * DET);

        // packs of the new h (shared by cand / pr1 heads), [h,ctx], [h,emb]
        packA(h_s, DET, nullptr, 0, nullptr, 0, KT_H, Ah);
        packA(h_s, DET, ctx_s, CTX, nullptr, 0, KT_HC, Ahc);
        packA(h_s, DET, emb_t, EMB, nullptr, 0, KT_HE, Ahe);

        // gated context update
        gemm(Ah,  KT_H,  Wcand_t, b_cand, cand_b, CTX, ACT_TANH);
        gemm(Ahc, KT_HC, Wgate_t, b_gate, o_gates + (long)t * B_ * CTX, CTX, ACT_SIG);
        ctx_update_kernel<<<cdiv((long)B_ * CTX, 256), 256, 0, stream>>>(
            o_gates + (long)t * B_ * CTX, cand_b, ctx_s, o_ctxs + (long)t * B_ * CTX);
        packA(ctx_s, CTX, nullptr, 0, nullptr, 0, KT_C, Ac);

        // prior head: elu(h @ W_pr1) @ W_pr2
        gemm(Ah, KT_H, Wpr1_t, b_pr1, tmp1_b, HID, ACT_ELU);
        packA(tmp1_b, HID, nullptr, 0, nullptr, 0, KT_T, At);
        gemm(At, KT_T, Wpr2_t, b_pr2, o_priors + (long)t * B_ * ZF, ZF, ACT_NONE);

        // context prior head: elu(ctx @ W_cp1) @ W_cp2
        gemm(Ac, KT_C, Wcp1_t, b_cp1, tmp1_b, HID, ACT_ELU);
        packA(tmp1_b, HID, nullptr, 0, nullptr, 0, KT_T, At);
        gemm(At, KT_T, Wcp2_t, b_cp2, o_cprior + (long)t * B_ * ZF, ZF, ACT_NONE);

        // posterior head: elu([h,emb] @ W_po1) @ W_po2
        gemm(Ahe, KT_HE, Wpo1_t, b_po1, tmp1_b, HID, ACT_ELU);
        packA(tmp1_b, HID, nullptr, 0, nullptr, 0, KT_T, At);
        gemm(At, KT_T, Wpo2_t, b_po2, o_posts + (long)t * B_ * ZF, ZF, ACT_NONE);

        // straight-through Gumbel categorical sample -> new z (one-hot forward)
        gumbel_sample_kernel<<<cdiv((long)B_ * 32, 256), 256, 0, stream>>>(
            o_posts + (long)t * B_ * ZF, z_s, o_stoch + (long)t * B_ * ZF, t);
    }

    // features = concat(deter, stoch, ctxs)
    features_kernel<<<cdiv(TB * (DET + ZF + CTX), 256), 256, 0, stream>>>(
        o_deter, o_stoch, o_ctxs, o_feat);

    // final carry
    hipMemcpyAsync(o_hf,    h_s,    (size_t)B_ * DET * 4, hipMemcpyDeviceToDevice, stream);
    hipMemcpyAsync(o_zf,    z_s,    (size_t)B_ * ZF  * 4, hipMemcpyDeviceToDevice, stream);
    hipMemcpyAsync(o_cf,    ctx_s,  (size_t)B_ * CTX * 4, hipMemcpyDeviceToDevice, stream);
    hipMemcpyAsync(o_cellf, cell_s, (size_t)B_ * DET * 4, hipMemcpyDeviceToDevice, stream);
}
